// Seq2Seq_90838558310723
// MI455X (gfx1250) — matrix operand order repaired and checked by reading, not compiled, not run on hardware
//
#include <hip/hip_runtime.h>

// ---------------------------------------------------------------------------
// Seq2Seq (bi-LSTM encoder + attention decoder) forward for MI455X (gfx1250).
//  * All GEMMs on v_wmma_f32_16x16x32_f16 (wave32 WMMA, f32 accumulate),
//    K-loop software-pipelined (ping-pong fragments) so VMEM overlaps WMMA.
//  * Weights converted f32->f16 once per call; Wo f16 = 164MB ~ fits 192MB L2.
//  * Attention: enc@Wa_e.T precomputed ONCE (removes ~540 GFLOPs vs naive).
//  * LSTM gates: one GEMM/step over concatenated [x|h] and [W_ih|W_hh].
// ---------------------------------------------------------------------------

typedef unsigned int u32;
typedef _Float16 h16;
typedef __attribute__((ext_vector_type(16))) _Float16 v16h;
typedef __attribute__((ext_vector_type(8)))  float    v8f;

#define S_LEN 128
#define T_LEN 64
#define BATCH 64
#define EMB   512
#define HID   512   // encoder hidden per direction
#define DEC   1024  // decoder hidden (=2H)
#define ATT   256
#define VOC   32000

union Frag16 { v16h v; u32 u[8]; };

__device__ __forceinline__ float sigf(float x) { return 1.0f / (1.0f + __expf(-x)); }

// A fragment: 16x32 f16 (M x K). lane -> row; khalf = lane>>4.
// VGPR v<4 : K = k0 + 2v + 8*lh ; VGPR v>=4 : K = k0 + 16 + 2(v-4) + 8*lh
__device__ __forceinline__ void load_a(const u32* __restrict__ xrow, int k0, int lh, Frag16& a) {
    const u32* xp = xrow + (k0 >> 1);
#pragma unroll
    for (int v = 0; v < 4; ++v) {
        a.u[v]     = xp[v + 4 * lh];
        a.u[4 + v] = xp[8 + v + 4 * lh];
    }
}
// B fragment: 32x16 (K x N). lane -> column; same K layout as A:
// VGPR v<4 : K = k0 + 2v + 8*lh ; VGPR v>=4 : K = k0 + 16 + 2(v-4) + 8*lh
__device__ __forceinline__ void load_b(const u32* __restrict__ wrow, int k0, int lh, Frag16& b) {
    const u32* wp = wrow + (k0 >> 1) + 4 * lh;
#pragma unroll
    for (int v = 0; v < 4; ++v) { b.u[v] = wp[v]; b.u[4 + v] = wp[8 + v]; }
}

__device__ __forceinline__ void mma4(const Frag16& a, const Frag16* b, v8f* acc) {
#pragma unroll
    for (int j = 0; j < 4; ++j)
        acc[j] = __builtin_amdgcn_wmma_f32_16x16x32_f16(
            false, a.v, false, b[j].v, (short)0, acc[j], false, false);
}

// ---------------------------------------------------------------------------
// Generic f16 WMMA GEMM:  C[M,N] = X[M,K] @ W[N,K]^T (+ bias[n] if mode==1)
// X row-major (stride ldx), W row-major (stride ldw), C f32 (stride ldc).
// Each wave: one 16x64 strip (4 n-subtiles share the A fragment).
// Requires: M%16==0, N%64==0, K%64==0, ldx/ldw even.
// ---------------------------------------------------------------------------
__global__ void __launch_bounds__(256)
gemm_h16_wmma(const h16* __restrict__ X, int ldx,
              const h16* __restrict__ W, int ldw,
              const float* __restrict__ bias,
              float* __restrict__ C, int ldc,
              int M, int N, int K, int mode)
{
    int wid  = (blockIdx.x * blockDim.x + threadIdx.x) >> 5;
    int lane = threadIdx.x & 31;
    int nT   = N >> 6;
    int total = (M >> 4) * nT;
    if (wid >= total) return;
    int mt = wid / nT, ng = wid % nT;
    int m0 = mt << 4, n0 = ng << 6;
    int lm = lane & 15, lh = lane >> 4;

    v8f acc[4];
#pragma unroll
    for (int j = 0; j < 4; ++j) {
        float binit = (mode == 1) ? bias[n0 + j * 16 + lm] : 0.0f;
#pragma unroll
        for (int r = 0; r < 8; ++r) acc[j][r] = binit;
    }

    const u32* xrow = (const u32*)(X + (size_t)(m0 + lm) * ldx);
    const u32* wrow[4];
#pragma unroll
    for (int j = 0; j < 4; ++j)
        wrow[j] = (const u32*)(W + (size_t)(n0 + j * 16 + lm) * ldw);

    // Software-pipelined K loop (2x unrolled, ping-pong fragments): loads for
    // the next 32-K chunk are issued before the WMMAs of the current chunk.
    Frag16 a0, a1, b0[4], b1[4];
    load_a(xrow, 0, lh, a0);
#pragma unroll
    for (int j = 0; j < 4; ++j) load_b(wrow[j], 0, lh, b0[j]);

    int k0 = 0;
    for (; k0 + 64 < K; k0 += 64) {
        load_a(xrow, k0 + 32, lh, a1);
#pragma unroll
        for (int j = 0; j < 4; ++j) load_b(wrow[j], k0 + 32, lh, b1[j]);
        mma4(a0, b0, acc);

        load_a(xrow, k0 + 64, lh, a0);
#pragma unroll
        for (int j = 0; j < 4; ++j) load_b(wrow[j], k0 + 64, lh, b0[j]);
        mma4(a1, b1, acc);
    }
    load_a(xrow, K - 32, lh, a1);
#pragma unroll
    for (int j = 0; j < 4; ++j) load_b(wrow[j], K - 32, lh, b1[j]);
    mma4(a0, b0, acc);
    mma4(a1, b1, acc);

#pragma unroll
    for (int j = 0; j < 4; ++j) {
        int n = n0 + j * 16 + lm;
#pragma unroll
        for (int r = 0; r < 8; ++r)
            C[(size_t)(m0 + r + 8 * lh) * ldc + n] = acc[j][r];
    }
}

// --------------------------- setup / conversion ---------------------------

__global__ void cvt_f32_h16(const float* __restrict__ s, h16* __restrict__ d, int n) {
    int i = blockIdx.x * 256 + threadIdx.x;
    if (i < n) d[i] = (h16)s[i];
}

// dst[row, 0:ka] = A[row,:], dst[row, ka:ka+kb] = B[row,:]  (f32->f16)
__global__ void cvt_cat2(const float* __restrict__ A, int ka,
                         const float* __restrict__ B, int kb,
                         h16* __restrict__ dst, int rows) {
    int i = blockIdx.x * 256 + threadIdx.x;
    int kk = ka + kb;
    if (i >= rows * kk) return;
    int r = i / kk, c = i - r * kk;
    float v = (c < ka) ? A[(size_t)r * ka + c] : B[(size_t)r * kb + (c - ka)];
    dst[i] = (h16)v;
}

__global__ void zero_f32(float* p, int n) {
    int i = blockIdx.x * 256 + threadIdx.x;
    if (i < n) p[i] = 0.0f;
}
__global__ void zero_h16(h16* p, int n) {
    int i = blockIdx.x * 256 + threadIdx.x;
    if (i < n) p[i] = (h16)0.0f;
}

// emb_h16[(s*B+b)*E + e] = enc_emb[src[s,b], e]
__global__ void enc_gather(const int* __restrict__ src, const float* __restrict__ tab,
                           h16* __restrict__ emb) {
    int i = blockIdx.x * 256 + threadIdx.x;   // S*B*E = 4194304
    if (i >= S_LEN * BATCH * EMB) return;
    int e = i & (EMB - 1);
    int rb = i >> 9;                          // s*B + b
    int tok = src[rb];
    emb[i] = (h16)tab[(size_t)tok * EMB + e];
}

// xh_f[b, 0:E] = emb[s],  xh_b[b, 0:E] = emb[S-1-s]
__global__ void enc_prep(const h16* __restrict__ emb, h16* __restrict__ xh_f,
                         h16* __restrict__ xh_b, int s) {
    int i = blockIdx.x * 256 + threadIdx.x;   // 2*B*E = 65536
    int dir = i >> 15;
    int r = i & 32767;
    int b = r >> 9, e = r & 511;
    int sp = dir ? (S_LEN - 1 - s) : s;
    h16* xh = dir ? xh_b : xh_f;
    xh[b * (EMB + HID) + e] = emb[((size_t)sp * BATCH + b) * EMB + e];
}

// Encoder LSTM cell for both directions. gates layout [B, 4H] = i,f,g,o.
__global__ void enc_cell(const float* __restrict__ gates_f, const float* __restrict__ gates_b,
                         float* __restrict__ c_f, float* __restrict__ c_b,
                         h16* __restrict__ xh_f, h16* __restrict__ xh_b,
                         h16* __restrict__ enc, h16* __restrict__ xd,
                         float* __restrict__ dec_c, int s) {
    int i = blockIdx.x * 256 + threadIdx.x;   // 2*B*H = 65536
    int dir = i >> 15;
    int r = i & 32767;
    int b = r >> 9, u = r & 511;
    const float* g = (dir ? gates_b : gates_f) + b * (4 * HID);
    float* c = (dir ? c_b : c_f);
    float ig = sigf(g[u]);
    float fg = sigf(g[HID + u]);
    float gg = tanhf(g[2 * HID + u]);
    float og = sigf(g[3 * HID + u]);
    float c2 = fg * c[b * HID + u] + ig * gg;
    float h2 = og * tanhf(c2);
    c[b * HID + u] = c2;
    (dir ? xh_b : xh_f)[b * (EMB + HID) + EMB + u] = (h16)h2;
    int spos = dir ? (S_LEN - 1 - s) : s;
    enc[((size_t)b * S_LEN + spos) * (2 * HID) + dir * HID + u] = (h16)h2;
    if (s == S_LEN - 1) {                      // seed decoder h0/c0
        xd[b * 2560 + 1536 + dir * HID + u] = (h16)h2;
        dec_c[b * DEC + dir * HID + u] = c2;
    }
}

// xd[b,0:E] = ybuf[b,2048:2560] = dec_emb[trg[t,b]]
__global__ void dec_gather(const int* __restrict__ trg, const float* __restrict__ tab,
                           h16* __restrict__ xd, h16* __restrict__ ybuf, int t) {
    int i = blockIdx.x * 256 + threadIdx.x;   // B*E = 32768
    if (i >= BATCH * EMB) return;
    int b = i >> 9, e = i & 511;
    int tok = trg[t * BATCH + b];
    h16 v = (h16)tab[(size_t)tok * EMB + e];
    xd[b * 2560 + e] = v;
    ybuf[b * 2560 + 2048 + e] = v;
}

// Per-step attention: scores (precomputed enc@Wa_e.T + per-step h@Wa_h.T),
// mask, softmax, context.  One block per batch element.
__global__ void __launch_bounds__(256)
att_kernel(const float* __restrict__ hWa, const float* __restrict__ att_pre,
           const float* __restrict__ vvec, const int* __restrict__ src,
           const h16* __restrict__ enc, h16* __restrict__ xd, h16* __restrict__ ybuf) {
    int b = blockIdx.x, tid = threadIdx.x;
    __shared__ float red[256];
    __shared__ float w[S_LEN];
    int s = tid >> 1, half = tid & 1;
    const float* ap = att_pre + ((size_t)b * S_LEN + s) * ATT + half * 128;
    const float* hw = hWa + b * ATT + half * 128;
    const float* vv = vvec + half * 128;
    float p = 0.0f;
#pragma unroll 4
    for (int a = 0; a < 128; ++a) p += vv[a] * tanhf(hw[a] + ap[a]);
    red[tid] = p;
    __syncthreads();
    if (half == 0) {
        float sc = red[tid] + red[tid + 1];
        if (src[s * BATCH + b] == 0) sc = -1e10f;   // PAD mask
        w[s] = sc;
    }
    __syncthreads();
    if (tid == 0) {                                  // softmax over 128 scores
        float mx = w[0];
        for (int i = 1; i < S_LEN; ++i) mx = fmaxf(mx, w[i]);
        float sum = 0.0f;
        for (int i = 0; i < S_LEN; ++i) { float e = __expf(w[i] - mx); w[i] = e; sum += e; }
        float inv = 1.0f / sum;
        for (int i = 0; i < S_LEN; ++i) w[i] *= inv;
    }
    __syncthreads();
    // ctx[b, e] = sum_s w[s] * enc[b, s, e],  e in [0, 2H)
#pragma unroll
    for (int j = 0; j < 4; ++j) {
        int e = tid + j * 256;
        float acc = 0.0f;
        const h16* ep = enc + (size_t)b * S_LEN * (2 * HID) + e;
        for (int s2 = 0; s2 < S_LEN; ++s2) acc += w[s2] * (float)ep[(size_t)s2 * (2 * HID)];
        xd[b * 2560 + EMB + e]  = (h16)acc;   // LSTM input ctx slot
        ybuf[b * 2560 + DEC + e] = (h16)acc;  // output-proj ctx slot
    }
}

// Decoder LSTM cell. gates [B, 4D].
__global__ void dec_cell(const float* __restrict__ gates, float* __restrict__ dec_c,
                         h16* __restrict__ xd, h16* __restrict__ ybuf) {
    int i = blockIdx.x * 256 + threadIdx.x;   // B*D = 65536
    int b = i >> 10, u = i & 1023;
    const float* g = gates + b * (4 * DEC);
    float ig = sigf(g[u]);
    float fg = sigf(g[DEC + u]);
    float gg = tanhf(g[2 * DEC + u]);
    float og = sigf(g[3 * DEC + u]);
    float c2 = fg * dec_c[i] + ig * gg;
    float h2 = og * tanhf(c2);
    dec_c[i] = c2;
    xd[b * 2560 + 1536 + u] = (h16)h2;   // h slot for next-step GEMMs
    ybuf[b * 2560 + u]      = (h16)h2;   // output-proj h slot
}

// ------------------------------- host side --------------------------------

static inline void launch_gemm(hipStream_t st, const h16* X, int ldx, const h16* W, int ldw,
                               const float* bias, float* C, int ldc,
                               int M, int N, int K, int mode) {
    int waves = (M >> 4) * (N >> 6);
    int blocks = (waves + 7) / 8;
    hipLaunchKernelGGL(gemm_h16_wmma, dim3(blocks), dim3(256), 0, st,
                       X, ldx, W, ldw, bias, C, ldc, M, N, K, mode);
}

extern "C" void kernel_launch(void* const* d_in, const int* in_sizes, int n_in,
                              void* d_out, int out_size, void* d_ws, size_t ws_size,
                              hipStream_t stream) {
    (void)in_sizes; (void)n_in; (void)out_size; (void)ws_size;
    const int*   src     = (const int*)  d_in[0];
    const int*   trg     = (const int*)  d_in[1];
    const float* enc_emb = (const float*)d_in[2];
    const float* Wf_ih   = (const float*)d_in[3];
    const float* Wf_hh   = (const float*)d_in[4];
    const float* bf      = (const float*)d_in[5];
    const float* Wb_ih   = (const float*)d_in[6];
    const float* Wb_hh   = (const float*)d_in[7];
    const float* bb      = (const float*)d_in[8];
    const float* Wa      = (const float*)d_in[9];
    const float* ba      = (const float*)d_in[10];
    const float* vvec    = (const float*)d_in[11];
    const float* dec_emb = (const float*)d_in[12];
    const float* Wd_ih   = (const float*)d_in[13];
    const float* Wd_hh   = (const float*)d_in[14];
    const float* bd      = (const float*)d_in[15];
    const float* Wo      = (const float*)d_in[16];
    const float* bo      = (const float*)d_in[17];
    float* out = (float*)d_out;

    size_t off = 0;
    auto carve = [&](size_t bytes) -> void* {
        void* p = (char*)d_ws + off;
        off += (bytes + 255) & ~(size_t)255;
        return p;
    };
    // f16 weights
    h16* Wcat_f = (h16*)carve((size_t)2048 * 1024 * 2);   // [Wf_ih | Wf_hh]
    h16* Wcat_b = (h16*)carve((size_t)2048 * 1024 * 2);
    h16* Wa16   = (h16*)carve((size_t)ATT * 2048 * 2);
    h16* Wdcat  = (h16*)carve((size_t)4096 * 2560 * 2);   // [Wd_ih | Wd_hh]
    h16* Wo16   = (h16*)carve((size_t)VOC * 2560 * 2);
    // activations
    h16* emb16  = (h16*)carve((size_t)S_LEN * BATCH * EMB * 2);
    h16* enc16  = (h16*)carve((size_t)BATCH * S_LEN * (2 * HID) * 2); // [B,S,2H]
    h16* xh_f   = (h16*)carve((size_t)BATCH * (EMB + HID) * 2);
    h16* xh_b   = (h16*)carve((size_t)BATCH * (EMB + HID) * 2);
    h16* xd     = (h16*)carve((size_t)BATCH * 2560 * 2);  // [emb|ctx|h]
    h16* ybuf   = (h16*)carve((size_t)BATCH * 2560 * 2);  // [h|ctx|emb]
    float* att_pre  = (float*)carve((size_t)BATCH * S_LEN * ATT * 4);
    float* gates_f  = (float*)carve((size_t)BATCH * 4 * HID * 4);
    float* gates_b  = (float*)carve((size_t)BATCH * 4 * HID * 4);
    float* dec_gate = (float*)carve((size_t)BATCH * 4 * DEC * 4);
    float* c_f      = (float*)carve((size_t)BATCH * HID * 4);
    float* c_b      = (float*)carve((size_t)BATCH * HID * 4);
    float* dec_c    = (float*)carve((size_t)BATCH * DEC * 4);
    float* hWa      = (float*)carve((size_t)BATCH * ATT * 4);

    #define NB(n) dim3(((n) + 255) / 256)

    // ---- weight conversion (f32 -> f16), once per call ----
    hipLaunchKernelGGL(cvt_cat2, NB(2048 * 1024), dim3(256), 0, stream,
                       Wf_ih, EMB, Wf_hh, HID, Wcat_f, 4 * HID);
    hipLaunchKernelGGL(cvt_cat2, NB(2048 * 1024), dim3(256), 0, stream,
                       Wb_ih, EMB, Wb_hh, HID, Wcat_b, 4 * HID);
    hipLaunchKernelGGL(cvt_f32_h16, NB(ATT * 2048), dim3(256), 0, stream,
                       Wa, Wa16, ATT * 2048);
    hipLaunchKernelGGL(cvt_cat2, NB(4096 * 2560), dim3(256), 0, stream,
                       Wd_ih, 1536, Wd_hh, DEC, Wdcat, 4 * DEC);
    hipLaunchKernelGGL(cvt_f32_h16, NB(VOC * 2560), dim3(256), 0, stream,
                       Wo, Wo16, VOC * 2560);

    // ---- init state / outputs[0] ----
    hipLaunchKernelGGL(zero_f32, NB(BATCH * HID), dim3(256), 0, stream, c_f, BATCH * HID);
    hipLaunchKernelGGL(zero_f32, NB(BATCH * HID), dim3(256), 0, stream, c_b, BATCH * HID);
    hipLaunchKernelGGL(zero_h16, NB(BATCH * (EMB + HID)), dim3(256), 0, stream, xh_f, BATCH * (EMB + HID));
    hipLaunchKernelGGL(zero_h16, NB(BATCH * (EMB + HID)), dim3(256), 0, stream, xh_b, BATCH * (EMB + HID));
    hipLaunchKernelGGL(zero_f32, NB(BATCH * VOC), dim3(256), 0, stream, out, BATCH * VOC);

    // ---- encoder embedding gather ----
    hipLaunchKernelGGL(enc_gather, NB(S_LEN * BATCH * EMB), dim3(256), 0, stream,
                       src, enc_emb, emb16);

    // ---- encoder: 128 sequential bi-LSTM steps ----
    for (int s = 0; s < S_LEN; ++s) {
        hipLaunchKernelGGL(enc_prep, NB(2 * BATCH * EMB), dim3(256), 0, stream,
                           emb16, xh_f, xh_b, s);
        launch_gemm(stream, xh_f, EMB + HID, Wcat_f, EMB + HID, bf,
                    gates_f, 4 * HID, BATCH, 4 * HID, EMB + HID, 1);
        launch_gemm(stream, xh_b, EMB + HID, Wcat_b, EMB + HID, bb,
                    gates_b, 4 * HID, BATCH, 4 * HID, EMB + HID, 1);
        hipLaunchKernelGGL(enc_cell, NB(2 * BATCH * HID), dim3(256), 0, stream,
                           gates_f, gates_b, c_f, c_b, xh_f, xh_b, enc16, xd, dec_c, s);
    }

    // ---- attention precompute: att_pre = enc @ Wa_e.T + ba  (once!) ----
    launch_gemm(stream, enc16, 2 * HID, Wa16 + DEC /*Wa[:,1024:]*/, 2048, ba,
                att_pre, ATT, BATCH * S_LEN, ATT, 2 * HID, 1);

    // ---- decoder: 63 sequential steps ----
    for (int t = 0; t < T_LEN - 1; ++t) {
        hipLaunchKernelGGL(dec_gather, NB(BATCH * EMB), dim3(256), 0, stream,
                           trg, dec_emb, xd, ybuf, t);
        // hWa = h @ Wa_h.T  (h lives at xd[:,1536:2560])
        launch_gemm(stream, xd + 1536, 2560, Wa16, 2048, nullptr,
                    hWa, ATT, BATCH, ATT, DEC, 0);
        hipLaunchKernelGGL(att_kernel, dim3(BATCH), dim3(256), 0, stream,
                           hWa, att_pre, vvec, src, enc16, xd, ybuf);
        // gates = [emb|ctx|h] @ [Wd_ih|Wd_hh].T + bd
        launch_gemm(stream, xd, 2560, Wdcat, 2560, bd,
                    dec_gate, 4 * DEC, BATCH, 4 * DEC, 2560, 1);
        hipLaunchKernelGGL(dec_cell, NB(BATCH * DEC), dim3(256), 0, stream,
                           dec_gate, dec_c, xd, ybuf);
        // outputs[t+1] = [h2|ctx|emb] @ Wo.T + bo
        launch_gemm(stream, ybuf, 2560, Wo16, 2560, bo,
                    out + (size_t)(t + 1) * BATCH * VOC, VOC,
                    BATCH, VOC, 2560, 1);
    }
    #undef NB
}
